// BinarizeConv2dSDP_26328149524873
// MI455X (gfx1250) — 8-bit matrix operand order repaired and checked by reading, compiled, not run on hardware
//
#include <hip/hip_runtime.h>
#include <hip/hip_bf16.h>

typedef __attribute__((ext_vector_type(8))) int   v8i;
typedef __attribute__((ext_vector_type(4))) int   v4i;
typedef __attribute__((ext_vector_type(2))) int   v2i;
typedef __attribute__((ext_vector_type(4))) float v4f;

typedef __attribute__((address_space(3))) v4i* lds_v4i_ptr;

#define IN_CHN   128
#define OUT_CHN  128
#define HW       56
#define NIMG     64
#define HW2      (HW*HW)
#define NPIX     (NIMG*HW2)
#define NWEIGHT  (OUT_CHN*IN_CHN*9)
#define NK_CHUNK 18
#define BFRAG_BYTES (NK_CHUNK*8*32*32)
#define BA_ELEMS (NPIX*IN_CHN)

#if __has_builtin(__builtin_amdgcn_global_load_async_to_lds_b128) && \
    __has_builtin(__builtin_amdgcn_s_wait_asynccnt)
#define USE_ASYNC_LDS 1
#else
#define USE_ASYNC_LDS 0
#endif

__global__ void __launch_bounds__(256) wgen_pack_kernel(
    const float* __restrict__ Mw, const float* __restrict__ Z,
    const float* __restrict__ rv, signed char* __restrict__ Bfrag) {
  int tid = blockIdx.x * 256 + threadIdx.x;
  if (tid >= BFRAG_BYTES) return;
  int b = tid & 3;
  int v = (tid >> 2) & 7;
  int l = (tid >> 5) & 31;
  int t = (tid >> 10) & 7;
  int c = tid >> 13;
  int half16 = ((l >> 4) & 1) * 16;
  int kk = (v < 4) ? (half16 + v * 4 + b)
                   : (32 + half16 + (v - 4) * 4 + b);
  int K  = c * 64 + kk;
  int tap = K >> 7;
  int ic  = K & 127;
  int kh = tap / 3, kw = tap - kh * 3;
  int oc = t * 16 + (l & 15);
  int flat = ((oc * IN_CHN + ic) * 3 + kh) * 3 + kw;
  float s = Mw[flat];
#pragma unroll
  for (int j = 0; j < 5; ++j)
    s = fmaf(rv[j], Z[j * NWEIGHT + flat], s);
  Bfrag[tid] = (signed char)((s > 0.f) - (s < 0.f));
}

__global__ void __launch_bounds__(256) binarize_kernel(
    const float* __restrict__ x, signed char* __restrict__ ba) {
  int tid = blockIdx.x * 256 + threadIdx.x;
  int w  = tid % HW;
  int t1 = tid / HW;
  int h  = t1 % HW;
  int t2 = t1 / HW;
  int ic = t2 & 127;
  int n  = t2 >> 7;
  float v = x[tid];
  ba[((n * HW + h) * HW + w) * IN_CHN + ic] = (signed char)((v > 0.f) - (v < 0.f));
}

__global__ void __launch_bounds__(256) conv_wmma_kernel(
    const signed char* __restrict__ ba, const signed char* __restrict__ Bfrag,
    const float* __restrict__ Alpha, float* __restrict__ out) {
  __shared__ signed char sB[BFRAG_BYTES];

  const int tid  = threadIdx.x;
  const int lane = tid & 31;
  const int wave = tid >> 5;

#if USE_ASYNC_LDS
#pragma unroll 4
  for (int i = 0; i < BFRAG_BYTES / (256 * 16); ++i) {
    const int off = (i * 256 + tid) * 16;
    __builtin_amdgcn_global_load_async_to_lds_b128(
        (v4i*)(Bfrag + off),
        (lds_v4i_ptr)(sB + off),
          0,   0);
  }
  __builtin_amdgcn_s_wait_asynccnt(0);
#else
#pragma unroll 4
  for (int i = 0; i < BFRAG_BYTES / (256 * 16); ++i) {
    const int off = (i * 256 + tid) * 16;
    *(v4i*)(sB + off) = *(const v4i*)(Bfrag + off);
  }
#endif
  __syncthreads();

  const int rowbase = blockIdx.x * 128 + wave * 16;
  const int r   = rowbase + (lane & 15);
  const int n   = r / HW2;
  const int rem = r - n * HW2;
  const int h   = rem / HW;
  const int w   = rem - h * HW;
  const int abase8 = ((lane >> 4) & 1) * 16;

  v8i acc[8];
#pragma unroll
  for (int t = 0; t < 8; ++t) acc[t] = v8i{};

#pragma unroll 1
  for (int c = 0; c < NK_CHUNK; ++c) {
    const int tap    = c >> 1;
    const int icbase = (c & 1) * 64;
    const int kh = tap / 3, kw = tap - kh * 3;
    const int hh = h + kh - 1, ww = w + kw - 1;

    v8i a = {};
    if (hh >= 0 && hh < HW && ww >= 0 && ww < HW) {
      const signed char* p =
          ba + (((n * HW + hh) * HW + ww) * IN_CHN + icbase + abase8);
      v2i d0 = *(const v2i*)(p +  0);
      v2i d1 = *(const v2i*)(p +  8);
      v2i d2 = *(const v2i*)(p + 32);
      v2i d3 = *(const v2i*)(p + 40);
      a[0] = d0.x; a[1] = d0.y; a[2] = d1.x; a[3] = d1.y;
      a[4] = d2.x; a[5] = d2.y; a[6] = d3.x; a[7] = d3.y;
    }

    const signed char* bp = sB + (c * 256 + lane) * 32;
#pragma unroll
    for (int t = 0; t < 8; ++t) {
      v4i b0 = *(const v4i*)(bp + t * 1024);
      v4i b1 = *(const v4i*)(bp + t * 1024 + 16);
      v8i bf;
      bf[0] = b0.x; bf[1] = b0.y; bf[2] = b0.z; bf[3] = b0.w;
      bf[4] = b1.x; bf[5] = b1.y; bf[6] = b1.z; bf[7] = b1.w;
      acc[t] = __builtin_amdgcn_wmma_i32_16x16x64_iu8(
           true, a,  true, bf, acc[t],
           false,  false);
    }
  }

  __syncthreads();
  float* sbuf = (float*)(sB + wave * 8192);

  const int mofs = (lane >> 4) ? 8 : 0;
#pragma unroll
  for (int t = 0; t < 8; ++t) {
    const int oc = t * 16 + (lane & 15);
    const float al = Alpha[oc];
#pragma unroll
    for (int j = 0; j < 8; ++j)
      sbuf[(mofs + j) * OUT_CHN + oc] = (float)acc[t][j] * al;
  }
  const int n0   = rowbase / HW2;
  const int rem0 = rowbase - n0 * HW2;
#pragma unroll
  for (int i = 0; i < 4; ++i) {
    const int oc2 = i * 32 + lane;
    float vals[16];
#pragma unroll
    for (int m = 0; m < 16; ++m) vals[m] = sbuf[m * OUT_CHN + oc2];
    float* dst = out + (n0 * OUT_CHN + oc2) * HW2 + rem0;
#pragma unroll
    for (int q = 0; q < 4; ++q) {
      v4f vv;
      vv.x = vals[q*4+0]; vv.y = vals[q*4+1];
      vv.z = vals[q*4+2]; vv.w = vals[q*4+3];
      *(v4f*)(dst + q * 4) = vv;
    }
  }
}

extern "C" void kernel_launch(void* const* d_in, const int* in_sizes, int n_in,
                              void* d_out, int out_size, void* d_ws, size_t ws_size,
                              hipStream_t stream) {
  const float* x     = (const float*)d_in[0];
  const float* Mw    = (const float*)d_in[1];
  const float* Z     = (const float*)d_in[2];
  const float* Alpha = (const float*)d_in[3];
  const float* rv    = (const float*)d_in[4];
  float* out = (float*)d_out;

  signed char* Bfrag = (signed char*)d_ws;
  signed char* ba    = (signed char*)d_ws + BFRAG_BYTES;

  wgen_pack_kernel<<<BFRAG_BYTES / 256, 256, 0, stream>>>(Mw, Z, rv, Bfrag);
  binarize_kernel<<<BA_ELEMS / 256, 256, 0, stream>>>(x, ba);
  conv_wmma_kernel<<<NPIX / 128, 256, 0, stream>>>(ba, Bfrag, Alpha, out);
}
